// AttentionBlock_37692632989791
// MI455X (gfx1250) — matrix operand order repaired and checked by reading, not compiled, not run on hardware
//
#include <hip/hip_runtime.h>

typedef __attribute__((ext_vector_type(16))) _Float16 v16h;
typedef __attribute__((ext_vector_type(8)))  _Float16 v8h;
typedef __attribute__((ext_vector_type(8)))  float    v8f;

union AFrag { v16h v; v8h h[2]; };

#define WMMA_F16(a,b,c) __builtin_amdgcn_wmma_f32_16x16x32_f16(false,(a),false,(b),(short)0,(c),false,false)

// ---------------- elementwise f32 -> f16 convert (weights) ----------------
__global__ void cvt_f16(const float* __restrict__ in, _Float16* __restrict__ out, int n) {
  int i = blockIdx.x * blockDim.x + threadIdx.x;
  if (i < n) out[i] = (_Float16)in[i];
}

// ---------------- GroupNorm (fp32 stats) + transpose to xT[b][t][c] f16 ----------------
__global__ void groupnorm_t(const float* __restrict__ x, const float* __restrict__ gamma,
                            const float* __restrict__ beta, _Float16* __restrict__ xT) {
  int b = blockIdx.x >> 5;       // 32 groups per batch
  int g = blockIdx.x & 31;
  const float* base = x + ((size_t)b * 512 + g * 16) * 1024;
  float s = 0.f, ss = 0.f;
  for (int i = threadIdx.x; i < 16 * 1024; i += 256) {
    float v = base[i]; s += v; ss += v * v;
  }
  __shared__ float rs[256], rss[256];
  rs[threadIdx.x] = s; rss[threadIdx.x] = ss; __syncthreads();
  for (int o = 128; o > 0; o >>= 1) {
    if (threadIdx.x < o) { rs[threadIdx.x] += rs[threadIdx.x + o]; rss[threadIdx.x] += rss[threadIdx.x + o]; }
    __syncthreads();
  }
  float mean = rs[0] * (1.f / 16384.f);
  float var  = rss[0] * (1.f / 16384.f) - mean * mean;
  float rstd = rsqrtf(var + 1e-5f);
  for (int i = threadIdx.x; i < 16 * 1024; i += 256) {
    int cl = i >> 10, t = i & 1023;
    int c = g * 16 + cl;
    float y = (base[i] - mean) * rstd * gamma[c] + beta[c];
    xT[((size_t)b * 1024 + t) * 512 + c] = (_Float16)y;
  }
}

// ---------------- transpose+convert attention_condition: [b][512][1024] -> acT[b][1024][512] ----------------
__global__ void transpose_cond(const float* __restrict__ ac, _Float16* __restrict__ acT) {
  __shared__ float tile[32][33];
  int b = blockIdx.z;
  int s0 = blockIdx.y * 32, t0 = blockIdx.x * 32;
  int tx = threadIdx.x, ty = threadIdx.y; // 32 x 8
  #pragma unroll
  for (int i = 0; i < 4; i++)
    tile[ty + 8 * i][tx] = ac[((size_t)b * 512 + s0 + ty + 8 * i) * 1024 + t0 + tx];
  __syncthreads();
  #pragma unroll
  for (int i = 0; i < 4; i++)
    acT[((size_t)b * 1024 + t0 + ty + 8 * i) * 512 + s0 + tx] = (_Float16)tile[tx][ty + 8 * i];
}

// ---------------- GEMM1: qkv = Wf(1536x512) @ x(512x1024) per batch, scatter to qT/kT/vB ----------------
__global__ void gemm_qkv(const _Float16* __restrict__ WfH, const _Float16* __restrict__ xT,
                         const float* __restrict__ bf,
                         _Float16* __restrict__ qT, _Float16* __restrict__ kT, _Float16* __restrict__ vB) {
  int wid  = blockIdx.x * 8 + (threadIdx.x >> 5);
  int lane = threadIdx.x & 31, ln = lane & 15, hi = lane >> 4;
  int nt = wid & 63; int tmp = wid >> 6; int mb = tmp % 24; int b = tmp / 24;
  int t0 = nt * 16, o0 = mb * 64;
  const _Float16* Bb = xT + ((size_t)b * 1024 + t0 + ln) * 512 + 16 * hi;
  v8f acc[4] = {};
  for (int k0 = 0; k0 < 512; k0 += 32) {
    v16h bfr = *(const v16h*)(Bb + k0);
    #pragma unroll
    for (int i = 0; i < 4; i++) {
      AFrag a;
      const _Float16* ap = WfH + (size_t)(o0 + i * 16 + ln) * 512 + k0;
      a.h[0] = *(const v8h*)(ap + 16 * hi);
      a.h[1] = *(const v8h*)(ap + 8 + 16 * hi);
      acc[i] = WMMA_F16(a.v, bfr, acc[i]);
    }
  }
  int bo = o0 / 64, h = bo / 3, reg = bo % 3, bh = b * 8 + h;
  int t = t0 + ln;
  #pragma unroll
  for (int i = 0; i < 4; i++) {
    #pragma unroll
    for (int r = 0; r < 8; r++) {
      int cl = i * 16 + r + 8 * hi;
      float val = acc[i][r] + bf[o0 + cl];
      if      (reg == 0) qT[((size_t)bh * 1024 + t) * 64 + cl] = (_Float16)val;
      else if (reg == 1) kT[((size_t)bh * 2048 + t) * 64 + cl] = (_Float16)val;
      else               vB[((size_t)bh * 64 + cl) * 2048 + t] = (_Float16)val;
    }
  }
}

// ---------------- GEMM2: cond = Wt(1024x512) @ ac(512x1024) per batch, scatter to kT/vB (s = 1024+t) ----------------
__global__ void gemm_cond(const _Float16* __restrict__ WtH, const _Float16* __restrict__ acT,
                          const float* __restrict__ bt,
                          _Float16* __restrict__ kT, _Float16* __restrict__ vB) {
  int wid  = blockIdx.x * 8 + (threadIdx.x >> 5);
  int lane = threadIdx.x & 31, ln = lane & 15, hi = lane >> 4;
  int nt = wid & 63; int tmp = wid >> 6; int mb = tmp % 16; int b = tmp / 16;
  int t0 = nt * 16, o0 = mb * 64;
  const _Float16* Bb = acT + ((size_t)b * 1024 + t0 + ln) * 512 + 16 * hi;
  v8f acc[4] = {};
  for (int k0 = 0; k0 < 512; k0 += 32) {
    v16h bfr = *(const v16h*)(Bb + k0);
    #pragma unroll
    for (int i = 0; i < 4; i++) {
      AFrag a;
      const _Float16* ap = WtH + (size_t)(o0 + i * 16 + ln) * 512 + k0;
      a.h[0] = *(const v8h*)(ap + 16 * hi);
      a.h[1] = *(const v8h*)(ap + 8 + 16 * hi);
      acc[i] = WMMA_F16(a.v, bfr, acc[i]);
    }
  }
  int bo = o0 / 64, h = bo / 2, reg = bo & 1, bh = b * 8 + h;
  int s = 1024 + t0 + ln;
  #pragma unroll
  for (int i = 0; i < 4; i++) {
    #pragma unroll
    for (int r = 0; r < 8; r++) {
      int cl = i * 16 + r + 8 * hi;
      float val = acc[i][r] + bt[o0 + cl];
      if (reg == 0) kT[((size_t)bh * 2048 + s) * 64 + cl] = (_Float16)val;
      else          vB[((size_t)bh * 64 + cl) * 2048 + s] = (_Float16)val;
    }
  }
}

// ---------------- attention: flash-style, scores computed transposed (s = rows) ----------------
__global__ void attention(const _Float16* __restrict__ qT, const _Float16* __restrict__ kT,
                          const _Float16* __restrict__ vB, _Float16* __restrict__ outT) {
  __shared__ alignas(64) _Float16 Plds[8][16][32];   // per-wave P^T bounce buffer
  int widb = threadIdx.x >> 5;
  int lane = threadIdx.x & 31, ln = lane & 15, hi = lane >> 4;
  int gwid = blockIdx.x * 8 + widb;
  int bh = gwid >> 6, t0 = (gwid & 63) * 16;

  const _Float16* qb = qT + ((size_t)bh * 1024 + t0 + ln) * 64;
  v16h bq0 = *(const v16h*)(qb + 16 * hi);        // c in [0,32)
  v16h bq1 = *(const v16h*)(qb + 32 + 16 * hi);   // c in [32,64)

  v8f acc[4] = {};                                 // O^T: 64(c) x 16(t)
  float m = -INFINITY, lsum = 0.f;
  _Float16* pl = &Plds[widb][0][0];                // [t][32] layout

  for (int s0 = 0; s0 < 2048; s0 += 32) {
    v8f l0 = {}, l1 = {};
    AFrag a;
    const _Float16* kp = kT + ((size_t)bh * 2048 + s0 + ln) * 64;
    a.h[0] = *(const v8h*)(kp + 16 * hi);      a.h[1] = *(const v8h*)(kp + 8 + 16 * hi);
    l0 = WMMA_F16(a.v, bq0, l0);
    a.h[0] = *(const v8h*)(kp + 32 + 16 * hi); a.h[1] = *(const v8h*)(kp + 40 + 16 * hi);
    l0 = WMMA_F16(a.v, bq1, l0);
    const _Float16* kp1 = kp + 16 * 64;
    a.h[0] = *(const v8h*)(kp1 + 16 * hi);      a.h[1] = *(const v8h*)(kp1 + 8 + 16 * hi);
    l1 = WMMA_F16(a.v, bq0, l1);
    a.h[0] = *(const v8h*)(kp1 + 32 + 16 * hi); a.h[1] = *(const v8h*)(kp1 + 40 + 16 * hi);
    l1 = WMMA_F16(a.v, bq1, l1);

    // online softmax over s (rows); stats are per-lane (lane <-> t)
    float mx = m;
    #pragma unroll
    for (int r = 0; r < 8; r++) {
      l0[r] *= 0.125f; l1[r] *= 0.125f;      // scale = HS^-0.5
      mx = fmaxf(mx, fmaxf(l0[r], l1[r]));
    }
    mx = fmaxf(mx, __shfl_xor(mx, 16, 32));
    float corr = __expf(m - mx);
    m = mx;
    float rsum = 0.f;
    #pragma unroll
    for (int r = 0; r < 8; r++) {
      float p0 = __expf(l0[r] - m), p1 = __expf(l1[r] - m);
      rsum += p0 + p1;
      pl[ln * 32 + r + 8 * hi]        = (_Float16)p0;   // P^T(s,t) -> [t][s]
      pl[ln * 32 + 16 + r + 8 * hi]   = (_Float16)p1;
    }
    rsum += __shfl_xor(rsum, 16, 32);
    lsum = lsum * corr + rsum;
    #pragma unroll
    for (int i = 0; i < 4; i++)
      #pragma unroll
      for (int r = 0; r < 8; r++) acc[i][r] *= corr;

    // O^T += V(c,s) @ P^T(s,t) ; same-wave DS ops are in-order, no barrier needed
    v16h bp = *(const v16h*)(pl + ln * 32 + 16 * hi);
    #pragma unroll
    for (int i = 0; i < 4; i++) {
      const _Float16* vp = vB + ((size_t)bh * 64 + i * 16 + ln) * 2048 + s0;
      a.h[0] = *(const v8h*)(vp + 16 * hi);
      a.h[1] = *(const v8h*)(vp + 8 + 16 * hi);
      acc[i] = WMMA_F16(a.v, bp, acc[i]);
    }
  }

  float inv = 1.f / lsum;
  int b = bh >> 3, h = bh & 7, t = t0 + ln;
  #pragma unroll
  for (int i = 0; i < 4; i++) {
    #pragma unroll
    for (int r = 0; r < 8; r++) {
      int cl = i * 16 + r + 8 * hi;
      outT[((size_t)b * 1024 + t) * 512 + h * 64 + cl] = (_Float16)(acc[i][r] * inv);
    }
  }
}

// ---------------- GEMM4: final = Wb(512x512) @ out(512x1024) + bb + residual ----------------
__global__ void gemm_back(const _Float16* __restrict__ WbH, const _Float16* __restrict__ outT,
                          const float* __restrict__ bb, const float* __restrict__ inF,
                          float* __restrict__ out) {
  int wid  = blockIdx.x * 8 + (threadIdx.x >> 5);
  int lane = threadIdx.x & 31, ln = lane & 15, hi = lane >> 4;
  int nt = wid & 63; int tmp = wid >> 6; int mb = tmp % 8; int b = tmp / 8;
  int t0 = nt * 16, o0 = mb * 64;
  const _Float16* Bb = outT + ((size_t)b * 1024 + t0 + ln) * 512 + 16 * hi;
  v8f acc[4] = {};
  for (int k0 = 0; k0 < 512; k0 += 32) {
    v16h bfr = *(const v16h*)(Bb + k0);
    #pragma unroll
    for (int i = 0; i < 4; i++) {
      AFrag a;
      const _Float16* ap = WbH + (size_t)(o0 + i * 16 + ln) * 512 + k0;
      a.h[0] = *(const v8h*)(ap + 16 * hi);
      a.h[1] = *(const v8h*)(ap + 8 + 16 * hi);
      acc[i] = WMMA_F16(a.v, bfr, acc[i]);
    }
  }
  int t = t0 + ln;
  #pragma unroll
  for (int i = 0; i < 4; i++) {
    #pragma unroll
    for (int r = 0; r < 8; r++) {
      int o = o0 + i * 16 + r + 8 * hi;
      size_t idx = ((size_t)b * 512 + o) * 1024 + t;
      out[idx] = acc[i][r] + bb[o] + inF[idx];
    }
  }
}

extern "C" void kernel_launch(void* const* d_in, const int* in_sizes, int n_in,
                              void* d_out, int out_size, void* d_ws, size_t ws_size,
                              hipStream_t stream) {
  const float* inF   = (const float*)d_in[0];
  const float* ac    = (const float*)d_in[1];
  const float* gamma = (const float*)d_in[2];
  const float* beta  = (const float*)d_in[3];
  const float* Wf    = (const float*)d_in[4];
  const float* bf    = (const float*)d_in[5];
  const float* Wt    = (const float*)d_in[6];
  const float* bt    = (const float*)d_in[7];
  const float* Wb    = (const float*)d_in[8];
  const float* bb    = (const float*)d_in[9];
  float* out = (float*)d_out;

  char* p = (char*)d_ws;
  auto alloc = [&](size_t halves) -> _Float16* {
    _Float16* r = (_Float16*)p;
    p += ((halves * 2 + 255) / 256) * 256;
    return r;
  };
  _Float16* xT   = alloc(8ull * 1024 * 512);   // GroupNorm output, [b][t][c]
  _Float16* acT  = alloc(8ull * 1024 * 512);   // condition transposed, [b][t][s]
  _Float16* WfH  = alloc(1536ull * 512);
  _Float16* WtH  = alloc(1024ull * 512);
  _Float16* WbH  = alloc(512ull * 512);
  _Float16* qT   = alloc(64ull * 1024 * 64);   // [bh][t][c]
  _Float16* kT   = alloc(64ull * 2048 * 64);   // [bh][s][c]
  _Float16* vB   = alloc(64ull * 64 * 2048);   // [bh][c][s]
  _Float16* outT = alloc(8ull * 1024 * 512);   // attention out, [b][t][c]

  cvt_f16<<<(1536 * 512 + 255) / 256, 256, 0, stream>>>(Wf, WfH, 1536 * 512);
  cvt_f16<<<(1024 * 512 + 255) / 256, 256, 0, stream>>>(Wt, WtH, 1024 * 512);
  cvt_f16<<<(512 * 512 + 255) / 256, 256, 0, stream>>>(Wb, WbH, 512 * 512);
  groupnorm_t<<<256, 256, 0, stream>>>(inF, gamma, beta, xT);
  transpose_cond<<<dim3(32, 16, 8), dim3(32, 8), 0, stream>>>(ac, acT);
  gemm_qkv <<<8 * 24 * 64 / 8, 256, 0, stream>>>(WfH, xT, bf, qT, kT, vB);
  gemm_cond<<<8 * 16 * 64 / 8, 256, 0, stream>>>(WtH, acT, bt, kT, vB);
  attention<<<64 * 64 / 8, 256, 0, stream>>>(qT, kT, vB, outT);
  gemm_back<<<8 * 8 * 64 / 8, 256, 0, stream>>>(WbH, outT, bb, inF, out);
}
